// AAGNN_multi_avg_66322884985285
// MI455X (gfx1250) — matrix operand order repaired and checked by reading, not compiled, not run on hardware
//
#include <hip/hip_runtime.h>

// ---------------------------------------------------------------------------
// GCN layer:  out = relu( (adj @ (x @ W)) * degree_norm / num_avg + b )
// N=16384, F=128, H=64.
// HBM traffic is dominated by the 1.07 GB adjacency matrix (read exactly
// once): floor ~46us @ 23.3 TB/s. The big GEMM runs on
// v_wmma_f32_16x16x32_bf16 with hi/lo bf16 splitting of both operands
// (3 WMMAs per K=32 step, al*bl ~2^-16 dropped) for near-f32 accuracy.
// Each wave computes 16 rows x 64 cols (1 A fragment -> 12 WMMAs/step) so
// the f32->bf16 conversion VALU is ~3.75 ops/WMMA and co-executes with the
// XDL pipe, keeping the kernel memory-bound.
// ---------------------------------------------------------------------------

#define N_NODES 16384
#define F_IN    128
#define H_OUT   64

typedef __attribute__((ext_vector_type(16))) __bf16        v16bf;
typedef __attribute__((ext_vector_type(8)))  float         v8f;
typedef __attribute__((ext_vector_type(4)))  float         f32x4;
typedef __attribute__((ext_vector_type(4)))  unsigned int  u32x4;

// ---------------------------------------------------------------------------
// Kernel 1: support = x @ W  (f32 FMA; tiny: 0.27 GFLOP), stored TRANSPOSED
// as two bf16 planes (hi, lo) so kernel 2's B-fragment loads are contiguous.
// supT_hi/lo layout: [H_OUT][N_NODES] row-major (stride N_NODES).
// ---------------------------------------------------------------------------
__global__ __launch_bounds__(256) void gcn_support_kernel(
    const float* __restrict__ x, const float* __restrict__ W,
    unsigned short* __restrict__ supT_hi, unsigned short* __restrict__ supT_lo) {
  __shared__ float xs[4][F_IN];
  const int tid = threadIdx.x;
  const int h   = tid & 63;       // output column 0..63
  const int rl  = tid >> 6;       // local row 0..3
  const int row = blockIdx.x * 4 + rl;

  // cooperative stage of 4 rows of x into LDS
  for (int i = tid; i < 4 * F_IN; i += 256)
    xs[i >> 7][i & 127] = x[(size_t)(blockIdx.x * 4 + (i >> 7)) * F_IN + (i & 127)];
  __syncthreads();

  float acc = 0.0f;
#pragma unroll 8
  for (int f = 0; f < F_IN; ++f)
    acc = __builtin_fmaf(xs[rl][f], W[f * H_OUT + h], acc);

  // f32 -> bf16 hi + lo split (RNE each step; residual ~2^-16 relative)
  __bf16 hi = (__bf16)acc;
  float  rem = acc - (float)hi;
  __bf16 lo = (__bf16)rem;
  union { __bf16 b; unsigned short u; } ch, cl;
  ch.b = hi; cl.b = lo;
  supT_hi[(size_t)h * N_NODES + row] = ch.u;
  supT_lo[(size_t)h * N_NODES + row] = cl.u;
}

// ---------------------------------------------------------------------------
// Kernel 2: agg = adj @ support via WMMA bf16, + degree/avg/bias/ReLU.
// Block = 256 threads = 8 waves; wave w owns rows [blk*128 + w*16, +16) and
// ALL 64 output columns (4 accumulators). Grid = 16384/128 = 128 blocks.
// A fragments loaded straight from global in the ISA 16-bit A layout (each
// adjacency element converted exactly once device-wide); B fragments from
// the transposed bf16 support planes (WGP$/L2 resident, shared by all waves
// of the block).
// ---------------------------------------------------------------------------
__global__ __launch_bounds__(256) void gcn_agg_kernel(
    const float* __restrict__ adj,
    const unsigned short* __restrict__ supT_hi,
    const unsigned short* __restrict__ supT_lo,
    const float* __restrict__ degree_norm,
    const float* __restrict__ bias,
    const int* __restrict__ num_avg_p,
    float* __restrict__ out) {
  const int tid   = threadIdx.x;
  const int lane  = tid & 31;
  const int wave  = tid >> 5;
  const int rb    = blockIdx.x * 128 + wave * 16;   // first row of this wave
  const int l15   = lane & 15;
  const bool lowhalf = (lane < 16);

  // A-fragment addressing (16-bit A 16x32 layout):
  //   lane<16 : row = rb + lane,    K runs {0..7}   and {16..23}
  //   lane>=16: row = rb + lane-16, K runs {8..15}  and {24..31}
  const int    rA   = rb + l15;
  const int    koff = lowhalf ? 0 : 8;
  const float* rowA = adj + (size_t)rA * N_NODES;

  // B-fragment addressing (B 32x16: lane = column; K half by half-wave):
  // per-lane element offset into supT planes for N-tile t is boff[t] + k0.
  const int kb = lowhalf ? 0 : 8;
  int boff[4];
#pragma unroll
  for (int t = 0; t < 4; ++t) boff[t] = (t * 16 + l15) * N_NODES + kb;

  v8f acc[4] = {};

  for (int k0 = 0; k0 < N_NODES; k0 += 32) {
    const float* pA = rowA + k0 + koff;
    // keep the adjacency stream ahead of the WMMA pipe
    __builtin_prefetch(pA + 256, 0, 1);

    f32x4 a0 = *(const f32x4*)(pA);
    f32x4 a1 = *(const f32x4*)(pA + 4);
    f32x4 a2 = *(const f32x4*)(pA + 16);
    f32x4 a3 = *(const f32x4*)(pA + 20);

    union BPack { u32x4 q[2]; v16bf v; } bh[4], bl[4];
#pragma unroll
    for (int t = 0; t < 4; ++t) {
      bh[t].q[0] = *(const u32x4*)(supT_hi + boff[t] + k0);
      bh[t].q[1] = *(const u32x4*)(supT_hi + boff[t] + k0 + 16);
      bl[t].q[0] = *(const u32x4*)(supT_lo + boff[t] + k0);
      bl[t].q[1] = *(const u32x4*)(supT_lo + boff[t] + k0 + 16);
    }

    v16bf ahi, alo;
    auto split = [&](float f, int e) {
      __bf16 h = (__bf16)f;
      ahi[e] = h;
      alo[e] = (__bf16)(f - (float)h);
    };
#pragma unroll
    for (int i = 0; i < 4; ++i) {
      split(a0[i], i);
      split(a1[i], 4 + i);
      split(a2[i], 8 + i);
      split(a3[i], 12 + i);
    }

    // a*b ~= ah*bh + ah*bl + al*bh   (al*bl ~ 2^-16, dropped)
#pragma unroll
    for (int t = 0; t < 4; ++t) {
      acc[t] = __builtin_amdgcn_wmma_f32_16x16x32_bf16(false, ahi, false, bh[t].v,
                                                       (short)0, acc[t], false, false);
      acc[t] = __builtin_amdgcn_wmma_f32_16x16x32_bf16(false, ahi, false, bl[t].v,
                                                       (short)0, acc[t], false, false);
      acc[t] = __builtin_amdgcn_wmma_f32_16x16x32_bf16(false, alo, false, bh[t].v,
                                                       (short)0, acc[t], false, false);
    }
  }

  // Epilogue: h = agg * degree_norm / num_avg + b ; relu.
  const float inv_avg = 1.0f / (float)num_avg_p[0];
  const int   mofs    = lowhalf ? 0 : 8;

  float dn[8];
#pragma unroll
  for (int v = 0; v < 8; ++v) dn[v] = degree_norm[rb + v + mofs];

#pragma unroll
  for (int t = 0; t < 4; ++t) {
    const float bcol = bias[t * 16 + l15];
#pragma unroll
    for (int v = 0; v < 8; ++v) {
      const int row = rb + v + mofs;                // C/D layout: M = v (+8 hi half)
      float r = acc[t][v] * dn[v] * inv_avg + bcol;
      r = fmaxf(r, 0.0f);
      out[(size_t)row * H_OUT + t * 16 + l15] = r;
    }
  }
}

// ---------------------------------------------------------------------------
extern "C" void kernel_launch(void* const* d_in, const int* in_sizes, int n_in,
                              void* d_out, int out_size, void* d_ws, size_t ws_size,
                              hipStream_t stream) {
  const float* x   = (const float*)d_in[0];   // [N, F]
  const float* adj = (const float*)d_in[1];   // [N, N]
  const float* dn  = (const float*)d_in[2];   // [N, 1]
  const float* W   = (const float*)d_in[3];   // [F, H]
  const float* b   = (const float*)d_in[4];   // [H]
  const int*   na  = (const int*)d_in[5];     // scalar num_avg

  unsigned short* supT_hi = (unsigned short*)d_ws;                       // 2 MB
  unsigned short* supT_lo = supT_hi + (size_t)H_OUT * N_NODES;           // 2 MB
  float* out = (float*)d_out;

  gcn_support_kernel<<<N_NODES / 4, 256, 0, stream>>>(x, W, supT_hi, supT_lo);
  gcn_agg_kernel<<<N_NODES / 32 / 4, 256, 0, stream>>>(adj, supT_hi, supT_lo,
                                                       dn, b, na, out);
}
